// HierarchicalMamba2_82386062672493
// MI455X (gfx1250) — matrix operand order repaired and checked by reading, not compiled, not run on hardware
//
#include <hip/hip_runtime.h>
#include <math.h>

#define D_MODEL   1024
#define D_INNER   2048
#define D_SSM     2048
#define D_STATE   128
#define D_CONVK   4
#define HEADDIM   64
#define NHEADS    32
#define CONV_DIM  2304
#define D_IN_PROJ 4384
#define BATCH     4
#define SEQLEN    2048
#define NTOK      (BATCH * SEQLEN)      /* 8192 */
#define DT_OFF    (D_INNER + CONV_DIM)  /* 4352 */

typedef __attribute__((ext_vector_type(16))) __bf16 bf16x16;
typedef __attribute__((ext_vector_type(8)))  __bf16 bf16x8;
typedef __attribute__((ext_vector_type(4)))  __bf16 bf16x4;
typedef __attribute__((ext_vector_type(8)))  float  floatx8;

union bv16 { bf16x16 v; bf16x8 h[2]; };

__device__ __forceinline__ bf16x8 ldb8(const __bf16* p) { return *(const bf16x8*)p; }
__device__ __forceinline__ float4 ldg4(const float* p)  { return *(const float4*)p; }

// ---------------------------------------------------------------------------
// fp32 -> (bf16 hi, bf16 residual) streaming split.  n % 1024 == 0.
// ---------------------------------------------------------------------------
__global__ __launch_bounds__(256) void cvt_split(
    const float* __restrict__ in, __bf16* __restrict__ hi,
    __bf16* __restrict__ lo, long n) {
  const long i4 = ((long)blockIdx.x * 256 + threadIdx.x) * 4;
  if (i4 >= n) return;
  float4 f = ldg4(in + i4);
  bf16x4 h, l;
  float ff[4] = { f.x, f.y, f.z, f.w };
#pragma unroll
  for (int e = 0; e < 4; ++e) {
    __bf16 hv = (__bf16)ff[e];
    h[e] = hv;
    l[e] = (__bf16)(ff[e] - (float)hv);
  }
  *(bf16x4*)(hi + i4) = h;
  *(bf16x4*)(lo + i4) = l;
}

// ---------------------------------------------------------------------------
// C[m,n] = sum_k A[m,k]*B[n,k], A/B pre-split into bf16 hi/lo (bf16x3 scheme).
// Wave tile 32x32 (2x2 WMMA), 4 waves/block -> block tile 128(M) x 32(N).
// Inner loop: 16x global_load_b128 + 12x v_wmma, no conversion VALU.
// ---------------------------------------------------------------------------
__global__ __launch_bounds__(128) void wmma_gemm_nt(
    const __bf16* __restrict__ Ahi, const __bf16* __restrict__ Alo,
    const __bf16* __restrict__ Bhi, const __bf16* __restrict__ Blo,
    float* __restrict__ C, int M, int N, int K) {
  const int lane = threadIdx.x & 31;
  const int wave = threadIdx.x >> 5;
  const int half = lane >> 4;
  const int r    = lane & 15;
  const int n0   = blockIdx.x * 32;
  const int m0   = blockIdx.y * 128 + wave * 32;

  floatx8 acc[2][2] = {};
  const __bf16* arh[2] = { Ahi + (size_t)(m0 + r) * K, Ahi + (size_t)(m0 + 16 + r) * K };
  const __bf16* arl[2] = { Alo + (size_t)(m0 + r) * K, Alo + (size_t)(m0 + 16 + r) * K };
  const __bf16* brh[2] = { Bhi + (size_t)(n0 + r) * K, Bhi + (size_t)(n0 + 16 + r) * K };
  const __bf16* brl[2] = { Blo + (size_t)(n0 + r) * K, Blo + (size_t)(n0 + 16 + r) * K };

  for (int kk = 0; kk < K; kk += 32) {
    bv16 ah[2], al[2], bh[2], bl[2];
    const int ka = kk + 8 * half;    // A: two 8-elem runs at ka, ka+16
    const int kb = kk + 8 * half;    // B: two 8-elem runs at kb, kb+16
#pragma unroll
    for (int i = 0; i < 2; ++i) {
      ah[i].h[0] = ldb8(arh[i] + ka);      ah[i].h[1] = ldb8(arh[i] + ka + 16);
      al[i].h[0] = ldb8(arl[i] + ka);      al[i].h[1] = ldb8(arl[i] + ka + 16);
    }
#pragma unroll
    for (int j = 0; j < 2; ++j) {
      bh[j].h[0] = ldb8(brh[j] + kb);      bh[j].h[1] = ldb8(brh[j] + kb + 16);
      bl[j].h[0] = ldb8(brl[j] + kb);      bl[j].h[1] = ldb8(brl[j] + kb + 16);
    }
#pragma unroll
    for (int i = 0; i < 2; ++i)
#pragma unroll
      for (int j = 0; j < 2; ++j) {
        acc[i][j] = __builtin_amdgcn_wmma_f32_16x16x32_bf16(
            false, ah[i].v, false, bh[j].v, (short)0, acc[i][j], false, false);
        acc[i][j] = __builtin_amdgcn_wmma_f32_16x16x32_bf16(
            false, al[i].v, false, bh[j].v, (short)0, acc[i][j], false, false);
        acc[i][j] = __builtin_amdgcn_wmma_f32_16x16x32_bf16(
            false, ah[i].v, false, bl[j].v, (short)0, acc[i][j], false, false);
      }
  }
  // C/D layout: lane (half, n=r); VGPR v -> (M = v + 8*half, N = r)
#pragma unroll
  for (int i = 0; i < 2; ++i)
#pragma unroll
    for (int j = 0; j < 2; ++j) {
      const int n = n0 + j * 16 + r;
#pragma unroll
      for (int v = 0; v < 8; ++v) {
        const int m = m0 + i * 16 + v + 8 * half;
        C[(size_t)m * N + n] = acc[i][j][v];
      }
    }
}

// ---------------------------------------------------------------------------
// causal depthwise conv1d (k=4) + SiLU on the xBC slice of zxbcdt
// ---------------------------------------------------------------------------
__global__ __launch_bounds__(256) void conv_silu(
    const float* __restrict__ zxbcdt, const float* __restrict__ cw,
    const float* __restrict__ cb, float* __restrict__ xbc) {
  const long idx = (long)blockIdx.x * 256 + threadIdx.x;
  if (idx >= (long)NTOK * CONV_DIM) return;
  const int  c   = (int)(idx % CONV_DIM);
  const long pos = idx / CONV_DIM;      // b*SEQLEN + l
  const int  l   = (int)(pos % SEQLEN);
  const long rowbase = pos * D_IN_PROJ + D_INNER;
  float acc = cb[c];
#pragma unroll
  for (int j = 0; j < D_CONVK; ++j) {
    const int dl = j - (D_CONVK - 1);   // -3..0
    if (l + dl >= 0)
      acc = fmaf(cw[c * D_CONVK + j], zxbcdt[rowbase + (long)dl * D_IN_PROJ + c], acc);
  }
  xbc[pos * CONV_DIM + c] = acc / (1.0f + __expf(-acc));
}

// ---------------------------------------------------------------------------
// selective scan: one workgroup per (head, batch); 64x128 state in registers.
// thread t: p = t>>2, owns n in [(t&3)*32, +32); quad-reduce via shfl_xor.
// ---------------------------------------------------------------------------
__global__ __launch_bounds__(256) void ssm_scan(
    const float* __restrict__ zxbcdt, const float* __restrict__ xbc,
    const float* __restrict__ dt_bias, const float* __restrict__ A_log,
    const float* __restrict__ Dp, float* __restrict__ y) {
  const int h = blockIdx.x;
  const int b = blockIdx.y;
  const int t = threadIdx.x;
  const int p  = t >> 2;
  const int nc = (t & 3) * 32;
  __shared__ float sBC[2][256];   // [0,128)=B, [128,256)=C, double buffered

  const float Ah = -(__expf(A_log[h]) + __expf(A_log[NHEADS + h]) +
                     __expf(A_log[2 * NHEADS + h])) * (1.0f / 3.0f);
  const float dtb = dt_bias[h];
  const float Dh  = Dp[h];

  float s[32];
#pragma unroll
  for (int i = 0; i < 32; ++i) s[i] = 0.0f;

  for (int l = 0; l < SEQLEN; ++l) {
    const long row = (long)b * SEQLEN + l;
    const int  buf = l & 1;
    sBC[buf][t] = xbc[row * CONV_DIM + D_SSM + t];   // B||C contiguous
    float dtr = zxbcdt[row * D_IN_PROJ + DT_OFF + h] + dtb;
    float dtv = (dtr > 20.0f) ? dtr : log1pf(__expf(dtr));   // softplus
    float dA  = __expf(dtv * Ah);
    float xp  = xbc[row * CONV_DIM + h * HEADDIM + p];
    __syncthreads();
    const float xdb = xp * dtv;
    float acc = 0.0f;
#pragma unroll
    for (int i = 0; i < 32; ++i) {
      const float Bv = sBC[buf][nc + i];
      const float Cv = sBC[buf][128 + nc + i];
      s[i] = fmaf(s[i], dA, xdb * Bv);
      acc  = fmaf(s[i], Cv, acc);
    }
    acc += __shfl_xor(acc, 1);
    acc += __shfl_xor(acc, 2);
    if ((t & 3) == 0)
      y[row * D_SSM + h * HEADDIM + p] = acc + Dh * xp;
  }
}

// ---------------------------------------------------------------------------
// RMSNorm over D_SSM, gate with silu(z); emit bf16 hi/lo directly for GEMM2.
// ---------------------------------------------------------------------------
__global__ __launch_bounds__(256) void rmsnorm_gate(
    const float* __restrict__ y, const float* __restrict__ zxbcdt,
    const float* __restrict__ nw, __bf16* __restrict__ ghi,
    __bf16* __restrict__ glo) {
  const long row = blockIdx.x;
  const float* yr = y + row * D_SSM;
  const float* zr = zxbcdt + row * D_IN_PROJ;   // z slice at offset 0
  __shared__ float red[256];
  float part = 0.0f;
  for (int i = threadIdx.x; i < D_SSM; i += 256) { float v = yr[i]; part = fmaf(v, v, part); }
  red[threadIdx.x] = part;
  __syncthreads();
  for (int st = 128; st > 0; st >>= 1) {
    if (threadIdx.x < st) red[threadIdx.x] += red[threadIdx.x + st];
    __syncthreads();
  }
  const float rstd = rsqrtf(red[0] * (1.0f / D_SSM) + 1e-5f);
  for (int i = threadIdx.x; i < D_SSM; i += 256) {
    const float z = zr[i];
    const float v = yr[i] * rstd * nw[i] * (z / (1.0f + __expf(-z)));
    __bf16 hv = (__bf16)v;
    ghi[row * D_SSM + i] = hv;
    glo[row * D_SSM + i] = (__bf16)(v - (float)hv);
  }
}

// ---------------------------------------------------------------------------
extern "C" void kernel_launch(void* const* d_in, const int* in_sizes, int n_in,
                              void* d_out, int out_size, void* d_ws, size_t ws_size,
                              hipStream_t stream) {
  (void)in_sizes; (void)n_in; (void)out_size; (void)ws_size;
  const float* u          = (const float*)d_in[0];
  const float* in_proj_w  = (const float*)d_in[1];
  const float* conv_w     = (const float*)d_in[2];
  const float* conv_b     = (const float*)d_in[3];
  const float* dt_bias    = (const float*)d_in[4];
  const float* A_log      = (const float*)d_in[5];
  const float* Dp         = (const float*)d_in[6];
  const float* norm_w     = (const float*)d_in[7];
  const float* out_proj_w = (const float*)d_in[8];
  float* out = (float*)d_out;

  char* ws = (char*)d_ws;
  const size_t SZ_ZX  = (size_t)NTOK * D_IN_PROJ * 4;   // 143,654,912
  const size_t SZ_U   = (size_t)NTOK * D_MODEL * 2;     //  16,777,216 (bf16)
  const size_t SZ_W1  = (size_t)D_IN_PROJ * D_MODEL * 2;//   8,978,432
  const size_t SZ_XBC = (size_t)NTOK * CONV_DIM * 4;    //  75,497,472
  const size_t SZ_YG  = (size_t)NTOK * D_SSM * 2;       //  33,554,432
  const size_t SZ_W2  = (size_t)D_MODEL * D_INNER * 2;  //   4,194,304

  float* zxbcdt = (float*)ws;
  char*  regB   = ws + SZ_ZX;                 // overlaid region (75.5 MB)
  float* yssm   = (float*)(ws + SZ_ZX + SZ_XBC);

  // Phase-1 occupants of regB: GEMM1 bf16 operands
  __bf16* u_hi  = (__bf16*)(regB);
  __bf16* u_lo  = (__bf16*)(regB + SZ_U);
  __bf16* w1_hi = (__bf16*)(regB + 2 * SZ_U);
  __bf16* w1_lo = (__bf16*)(regB + 2 * SZ_U + SZ_W1);
  // Phase-2 occupant of regB: conv output
  float*  xbc   = (float*)regB;
  // Phase-3 occupants of regB: GEMM2 bf16 operands (exactly fills 75.5 MB)
  __bf16* yg_hi = (__bf16*)(regB);
  __bf16* yg_lo = (__bf16*)(regB + SZ_YG);
  __bf16* w2_hi = (__bf16*)(regB + 2 * SZ_YG);
  __bf16* w2_lo = (__bf16*)(regB + 2 * SZ_YG + SZ_W2);

  // 0) split GEMM1 operands into bf16 hi/lo (streaming, memory bound)
  {
    long nU = (long)NTOK * D_MODEL;           // 8,388,608
    long nW = (long)D_IN_PROJ * D_MODEL;      // 4,489,216
    cvt_split<<<(unsigned)(nU / 4 / 256), 256, 0, stream>>>(u, u_hi, u_lo, nU);
    cvt_split<<<(unsigned)(nW / 4 / 256), 256, 0, stream>>>(in_proj_w, w1_hi, w1_lo, nW);
  }

  // 1) zxbcdt = u @ in_proj_w^T   (M=8192, N=4384, K=1024)
  dim3 g1(D_IN_PROJ / 32, NTOK / 128);
  wmma_gemm_nt<<<g1, 128, 0, stream>>>(u_hi, u_lo, w1_hi, w1_lo, zxbcdt,
                                       NTOK, D_IN_PROJ, D_MODEL);

  // 2) causal depthwise conv + SiLU   (regB: GEMM1 operands dead -> xbc)
  conv_silu<<<(NTOK * CONV_DIM) / 256, 256, 0, stream>>>(zxbcdt, conv_w, conv_b, xbc);

  // 3) selective scan (+ D*x skip)
  dim3 gs(NHEADS, BATCH);
  ssm_scan<<<gs, 256, 0, stream>>>(zxbcdt, xbc, dt_bias, A_log, Dp, yssm);

  // 4) split out_proj_w, then RMSNorm+gate -> bf16 hi/lo (xbc dead now)
  {
    long nW2 = (long)D_MODEL * D_INNER;       // 2,097,152
    cvt_split<<<(unsigned)(nW2 / 4 / 256), 256, 0, stream>>>(out_proj_w, w2_hi, w2_lo, nW2);
  }
  rmsnorm_gate<<<NTOK, 256, 0, stream>>>(yssm, zxbcdt, norm_w, yg_hi, yg_lo);

  // 5) out = ygated @ out_proj_w^T   (M=8192, N=1024, K=2048)
  dim3 g2(D_MODEL / 32, NTOK / 128);
  wmma_gemm_nt<<<g2, 128, 0, stream>>>(yg_hi, yg_lo, w2_hi, w2_lo, out,
                                       NTOK, D_MODEL, D_INNER);
}
